// FlatQuantPixArtFeedForward_42717744726517
// MI455X (gfx1250) — compile-verified, matrix operand order checked by reading, not run on hardware
//
#include <hip/hip_runtime.h>
#include <hip/hip_bf16.h>

// ---------------------------------------------------------------------------
// FlatQuant PixArt FFN on MI455X (gfx1250, wave32, WMMA).
// Heavy math (projections AND kronecker transforms via materialized Kronecker
// matrices) runs through v_wmma_f32_16x16x32_f16. GEMM tiles are staged with
// gfx1250 async global->LDS copies (ASYNCcnt) and double buffering, with a
// peeled steady-state loop (unconditional next-tile issue, s_wait_asynccnt 3).
// ---------------------------------------------------------------------------

#define NTOK 16384          // B*S = 4*4096
#define DIM  1152
#define DFF  4608
#define UL_N 32
#define UR_N 36
#define DL_N 64
#define DR_N 72

typedef __attribute__((ext_vector_type(16))) _Float16 v16h;
typedef __attribute__((ext_vector_type(8)))  float    v8f;

union Frag { v16h h; uint4 u[2]; };

__device__ __forceinline__ float gelu_tanh(float x) {
    float x3 = x * x * x;
    return 0.5f * x * (1.0f + tanhf(0.7978845608028654f * (x + 0.044715f * x3)));
}

// ---------------------------------------------------------------------------
// Gauss-Jordan inverse of an n x n near-identity matrix (n <= 72), one block.
// ---------------------------------------------------------------------------
__global__ void invert_kernel(const float* __restrict__ A, float* __restrict__ out, int n) {
    extern __shared__ float gj[];           // n x 2n augmented [A | I]
    const int t = threadIdx.x;
    const int w = 2 * n;
    for (int idx = t; idx < n * w; idx += blockDim.x) {
        int r = idx / w, c = idx % w;
        gj[idx] = (c < n) ? A[r * n + c] : ((c - n == r) ? 1.0f : 0.0f);
    }
    __syncthreads();
    for (int k = 0; k < n; ++k) {
        float piv = gj[k * w + k];
        __syncthreads();
        float inv = 1.0f / piv;
        for (int j = t; j < w; j += blockDim.x) gj[k * w + j] *= inv;
        __syncthreads();
        for (int i = t; i < n; i += blockDim.x) {
            if (i != k) {
                float f = gj[i * w + k];
                for (int j = 0; j < w; ++j) gj[i * w + j] -= f * gj[k * w + j];
            }
        }
        __syncthreads();
    }
    for (int idx = t; idx < n * n; idx += blockDim.x) {
        int r = idx / n, c = idx % n;
        out[idx] = gj[r * w + n + c];
    }
}

// ---------------------------------------------------------------------------
// Materialize transposed Kronecker matrix, fp16, [dout][din] row-major.
// mode 0 (activation):  K[din,dout] = diag[din] * L[l,m] * R[r,k]
// mode 1 (weight, L/R are the inverses): K[din,dout] = (1/diag[din]) * invL[m,l] * invR[k,r]
//   where din = l*Rsz + r, dout = m*Rsz + k.
// ---------------------------------------------------------------------------
__global__ void build_kron(_Float16* __restrict__ outT,
                           const float* __restrict__ L, const float* __restrict__ R,
                           const float* __restrict__ diag,
                           int Lsz, int Rsz, int mode) {
    const long D = (long)Lsz * Rsz;
    const long total = D * D;
    long idx = (long)blockIdx.x * blockDim.x + threadIdx.x;
    if (idx >= total) return;
    int dout = (int)(idx / D), din = (int)(idx % D);
    int l = din / Rsz, r = din % Rsz;
    int m = dout / Rsz, k = dout % Rsz;
    float v;
    if (mode == 0) v = diag[din] * L[l * Lsz + m] * R[r * Rsz + k];
    else           v = (1.0f / diag[din]) * L[m * Lsz + l] * R[k * Rsz + r];
    outT[(long)dout * D + din] = (_Float16)v;
}

// ---------------------------------------------------------------------------
// fp32 -> fp16 convert
// ---------------------------------------------------------------------------
__global__ void f32_to_f16(const float* __restrict__ in, _Float16* __restrict__ out, long n) {
    long i = (long)blockIdx.x * blockDim.x + threadIdx.x;
    if (i < n) out[i] = (_Float16)in[i];
}

// ---------------------------------------------------------------------------
// In-place rowwise asymmetric 4-bit fake quant (per token), block per row.
// ---------------------------------------------------------------------------
__global__ __launch_bounds__(256) void quant_act_rows(_Float16* __restrict__ buf, int rowlen) {
    __shared__ float smin[256], smax[256];
    const int t = threadIdx.x;
    _Float16* p = buf + (long)blockIdx.x * rowlen;
    float vmin = 1e30f, vmax = -1e30f;
    for (int j = t; j < rowlen; j += 256) {
        float x = (float)p[j];
        vmin = fminf(vmin, x); vmax = fmaxf(vmax, x);
    }
    smin[t] = vmin; smax[t] = vmax;
    __syncthreads();
    for (int s = 128; s > 0; s >>= 1) {
        if (t < s) { smin[t] = fminf(smin[t], smin[t + s]); smax[t] = fmaxf(smax[t], smax[t + s]); }
        __syncthreads();
    }
    float scale = fmaxf(smax[0] - smin[0], 1e-8f) / 15.0f;
    float zp    = rintf(-smin[0] / scale);
    for (int j = t; j < rowlen; j += 256) {
        float x = (float)p[j];
        float q = fminf(fmaxf(rintf(x / scale) + zp, 0.0f), 15.0f);
        p[j] = (_Float16)((q - zp) * scale);
    }
}

// ---------------------------------------------------------------------------
// In-place rowwise symmetric 4-bit fake quant (per output channel).
// ---------------------------------------------------------------------------
__global__ __launch_bounds__(256) void quant_weight_rows(_Float16* __restrict__ buf, int rowlen) {
    __shared__ float samax[256];
    const int t = threadIdx.x;
    _Float16* p = buf + (long)blockIdx.x * rowlen;
    float amax = 0.0f;
    for (int j = t; j < rowlen; j += 256) amax = fmaxf(amax, fabsf((float)p[j]));
    samax[t] = amax;
    __syncthreads();
    for (int s = 128; s > 0; s >>= 1) {
        if (t < s) samax[t] = fmaxf(samax[t], samax[t + s]);
        __syncthreads();
    }
    float scale = fmaxf(samax[0], 1e-8f) / 7.0f;
    for (int j = t; j < rowlen; j += 256) {
        float q = fminf(fmaxf(rintf((float)p[j] / scale), -8.0f), 7.0f);
        p[j] = (_Float16)(q * scale);
    }
}

// ---------------------------------------------------------------------------
// NT GEMM: C[M,N] = A[M,K] * B[N,K]^T   (fp16 in, f32 accumulate via WMMA)
// Block: 256 threads = 8 waves, tile 128x64, waves in 4x2 grid, 2x2 WMMA
// tiles per wave. K step 32. Tiles staged with gfx1250 async global->LDS
// copies, double buffered: steady-state iterations issue the next tile
// unconditionally then s_wait_asynccnt 3 (retires previous tile, next stays
// in flight); peeled tail waits to 0.
// MODE: 0 = store f16; 1 = f16 gelu(x+bias); 2 = f32 x+bias.
// ---------------------------------------------------------------------------
template <int MODE>
__global__ __launch_bounds__(256) void gemm_nt(const _Float16* __restrict__ A,
                                               const _Float16* __restrict__ B,
                                               void* __restrict__ Cout,
                                               const float* __restrict__ bias,
                                               int M, int N, int K) {
    constexpr int LDSS = 40;                       // padded row stride (halves)
    __shared__ alignas(16) _Float16 As[2][128 * LDSS];
    __shared__ alignas(16) _Float16 Bs[2][64 * LDSS];

    const int t    = threadIdx.x;
    const int lane = t & 31;
    const int wave = t >> 5;
    const int wm   = wave & 3;                     // 0..3 -> M
    const int wn   = wave >> 2;                    // 0..1 -> N
    const long blockRow = (long)blockIdx.y * 128;
    const long blockCol = (long)blockIdx.x * 64;

    // async staging assignment: 8-half (16B) chunks, advancing pointers by K-step
    const int rq = t >> 2;                         // 0..63 : row
    const int q  = t & 3;                          // 0..3  : 8-half chunk in 32-col tile
    const _Float16* gA0 = A + (blockRow + rq)      * (long)K + q * 8;
    const _Float16* gA1 = A + (blockRow + rq + 64) * (long)K + q * 8;
    const _Float16* gB  = B + (blockCol + rq)      * (long)K + q * 8;

    const unsigned laA0 = (unsigned)(uintptr_t)&As[0][rq * LDSS + q * 8];
    const unsigned laA1 = laA0 + 64u * LDSS * 2u;
    const unsigned laB  = (unsigned)(uintptr_t)&Bs[0][rq * LDSS + q * 8];
    constexpr unsigned strA = sizeof(_Float16) * 128 * LDSS;
    constexpr unsigned strB = sizeof(_Float16) * 64 * LDSS;

    auto issue_tile = [&](int b) {                 // issues current gA*/gB, then advances
        unsigned oa = (unsigned)b * strA;
        unsigned ob = (unsigned)b * strB;
        asm volatile("global_load_async_to_lds_b128 %0, %1, off"
                     :: "v"(laA0 + oa), "v"(gA0) : "memory");
        asm volatile("global_load_async_to_lds_b128 %0, %1, off"
                     :: "v"(laA1 + oa), "v"(gA1) : "memory");
        asm volatile("global_load_async_to_lds_b128 %0, %1, off"
                     :: "v"(laB + ob),  "v"(gB)  : "memory");
        gA0 += 32; gA1 += 32; gB += 32;
    };

    v8f acc[2][2] = {};

    const int hs = lane >> 4;                      // K-half select (ISA A/B layout)
    const int rr = lane & 15;                      // row-in-tile

    auto mac_step = [&](int b) {
        const _Float16* Ab = As[b];
        const _Float16* Bb = Bs[b];
        Frag af[2], bf[2];
#pragma unroll
        for (int mi = 0; mi < 2; ++mi) {
            int row = wm * 32 + mi * 16 + rr;
            af[mi].u[0] = *(const uint4*)&Ab[row * LDSS + hs * 8];
            af[mi].u[1] = *(const uint4*)&Ab[row * LDSS + 16 + hs * 8];
        }
#pragma unroll
        for (int ni = 0; ni < 2; ++ni) {
            int col = wn * 32 + ni * 16 + rr;
            bf[ni].u[0] = *(const uint4*)&Bb[col * LDSS + hs * 8];
            bf[ni].u[1] = *(const uint4*)&Bb[col * LDSS + 16 + hs * 8];
        }
#pragma unroll
        for (int mi = 0; mi < 2; ++mi)
#pragma unroll
            for (int ni = 0; ni < 2; ++ni)
                acc[mi][ni] = __builtin_amdgcn_wmma_f32_16x16x32_f16(
                    false, af[mi].h, false, bf[ni].h,
                    (short)0, acc[mi][ni], false, false);
    };

    const int nIter = K >> 5;
    issue_tile(0);
    int buf = 0;
    for (int i = 0; i < nIter - 1; ++i) {          // steady state: always a next tile
        issue_tile(buf ^ 1);
        asm volatile("s_wait_asynccnt 0x3" ::: "memory");
        __syncthreads();                           // all waves' tile resident
        mac_step(buf);
        __syncthreads();                           // safe to overwrite buf next iter
        buf ^= 1;
    }
    asm volatile("s_wait_asynccnt 0x0" ::: "memory");
    __syncthreads();
    mac_step(buf);

    // Epilogue. C layout: VGPR r -> M = r + (lane>=16)*8 ; N = lane & 15.
#pragma unroll
    for (int mi = 0; mi < 2; ++mi)
#pragma unroll
        for (int ni = 0; ni < 2; ++ni) {
            long mbase = blockRow + wm * 32 + mi * 16 + hs * 8;
            long n     = blockCol + wn * 32 + ni * 16 + rr;
#pragma unroll
            for (int r = 0; r < 8; ++r) {
                long  m = mbase + r;
                float v = acc[mi][ni][r];
                if constexpr (MODE == 2) {
                    ((float*)Cout)[m * (long)N + n] = v + bias[n];
                } else {
                    if constexpr (MODE == 1) v = gelu_tanh(v + bias[n]);
                    ((_Float16*)Cout)[m * (long)N + n] = (_Float16)v;
                }
            }
        }
}

// ---------------------------------------------------------------------------
// Host orchestration
// ---------------------------------------------------------------------------
extern "C" void kernel_launch(void* const* d_in, const int* in_sizes, int n_in,
                              void* d_out, int out_size, void* d_ws, size_t ws_size,
                              hipStream_t stream) {
    const float* x       = (const float*)d_in[0];
    const float* up_w    = (const float*)d_in[1];
    const float* up_b    = (const float*)d_in[2];
    const float* down_w  = (const float*)d_in[3];
    const float* down_b  = (const float*)d_in[4];
    const float* up_l    = (const float*)d_in[5];
    const float* up_r    = (const float*)d_in[6];
    const float* up_d    = (const float*)d_in[7];
    const float* dn_l    = (const float*)d_in[8];
    const float* dn_r    = (const float*)d_in[9];
    const float* dn_d    = (const float*)d_in[10];

    // bump allocator over workspace
    size_t off = 0;
    auto wsAlloc = [&](size_t bytes) -> char* {
        char* p = (char*)d_ws + off;
        off += (bytes + 255) & ~(size_t)255;
        return p;
    };

    float* inv_ul = (float*)wsAlloc(UL_N * UL_N * 4);
    float* inv_ur = (float*)wsAlloc(UR_N * UR_N * 4);
    float* inv_dl = (float*)wsAlloc(DL_N * DL_N * 4);
    float* inv_dr = (float*)wsAlloc(DR_N * DR_N * 4);

    _Float16* KaUpT = (_Float16*)wsAlloc((size_t)DIM * DIM * 2);
    _Float16* KwUpT = (_Float16*)wsAlloc((size_t)DIM * DIM * 2);
    _Float16* KaDnT = (_Float16*)wsAlloc((size_t)DFF * DFF * 2);
    _Float16* KwDnT = (_Float16*)wsAlloc((size_t)DFF * DFF * 2);

    _Float16* xh   = (_Float16*)wsAlloc((size_t)NTOK * DIM * 2);
    _Float16* Aup  = (_Float16*)wsAlloc((size_t)NTOK * DIM * 2);
    _Float16* wupH = (_Float16*)wsAlloc((size_t)DFF * DIM * 2);
    _Float16* Wup  = (_Float16*)wsAlloc((size_t)DFF * DIM * 2);
    _Float16* Hg   = (_Float16*)wsAlloc((size_t)NTOK * DFF * 2);
    _Float16* Adn  = (_Float16*)wsAlloc((size_t)NTOK * DFF * 2);
    _Float16* wdnH = (_Float16*)wsAlloc((size_t)DIM * DFF * 2);
    _Float16* Wdn  = (_Float16*)wsAlloc((size_t)DIM * DFF * 2);

    // 1) small matrix inverses (Gauss-Jordan in LDS)
    invert_kernel<<<1, 128, 2 * UL_N * UL_N * 4, stream>>>(up_l, inv_ul, UL_N);
    invert_kernel<<<1, 128, 2 * UR_N * UR_N * 4, stream>>>(up_r, inv_ur, UR_N);
    invert_kernel<<<1, 128, 2 * DL_N * DL_N * 4, stream>>>(dn_l, inv_dl, DL_N);
    invert_kernel<<<1, 128, 2 * DR_N * DR_N * 4, stream>>>(dn_r, inv_dr, DR_N);

    // 2) materialize transposed Kronecker matrices (diag folded in)
    {
        long tu = (long)DIM * DIM, td = (long)DFF * DFF;
        build_kron<<<(tu + 255) / 256, 256, 0, stream>>>(KaUpT, up_l,   up_r,   up_d, UL_N, UR_N, 0);
        build_kron<<<(tu + 255) / 256, 256, 0, stream>>>(KwUpT, inv_ul, inv_ur, up_d, UL_N, UR_N, 1);
        build_kron<<<(td + 255) / 256, 256, 0, stream>>>(KaDnT, dn_l,   dn_r,   dn_d, DL_N, DR_N, 0);
        build_kron<<<(td + 255) / 256, 256, 0, stream>>>(KwDnT, inv_dl, inv_dr, dn_d, DL_N, DR_N, 1);
    }

    // 3) fp16 copies of x and weights
    {
        long n1 = (long)NTOK * DIM, n2 = (long)DFF * DIM, n3 = (long)DIM * DFF;
        f32_to_f16<<<(n1 + 255) / 256, 256, 0, stream>>>(x,      xh,   n1);
        f32_to_f16<<<(n2 + 255) / 256, 256, 0, stream>>>(up_w,   wupH, n2);
        f32_to_f16<<<(n3 + 255) / 256, 256, 0, stream>>>(down_w, wdnH, n3);
    }

    // 4) up activation transform (kron as GEMM) + per-token quant (in place)
    gemm_nt<0><<<dim3(DIM / 64, NTOK / 128), 256, 0, stream>>>(xh, KaUpT, Aup, nullptr, NTOK, DIM, DIM);
    quant_act_rows<<<NTOK, 256, 0, stream>>>(Aup, DIM);

    // 5) up weight transform + per-row quant (in place)
    gemm_nt<0><<<dim3(DIM / 64, DFF / 128), 256, 0, stream>>>(wupH, KwUpT, Wup, nullptr, DFF, DIM, DIM);
    quant_weight_rows<<<DFF, 256, 0, stream>>>(Wup, DIM);

    // 6) up projection + bias + tanh-GELU
    gemm_nt<1><<<dim3(DFF / 64, NTOK / 128), 256, 0, stream>>>(Aup, Wup, Hg, up_b, NTOK, DFF, DIM);

    // 7) down activation transform + per-token quant
    gemm_nt<0><<<dim3(DFF / 64, NTOK / 128), 256, 0, stream>>>(Hg, KaDnT, Adn, nullptr, NTOK, DFF, DFF);
    quant_act_rows<<<NTOK, 256, 0, stream>>>(Adn, DFF);

    // 8) down weight transform + per-row quant
    gemm_nt<0><<<dim3(DFF / 64, DIM / 128), 256, 0, stream>>>(wdnH, KwDnT, Wdn, nullptr, DIM, DFF, DFF);
    quant_weight_rows<<<DIM, 256, 0, stream>>>(Wdn, DFF);

    // 9) down projection + bias -> fp32 output
    gemm_nt<2><<<dim3(DIM / 64, NTOK / 128), 256, 0, stream>>>(Adn, Wdn, d_out, down_b, NTOK, DIM, DFF);
}
